// Attention_45930380263558
// MI455X (gfx1250) — matrix operand order repaired and checked by reading, not compiled, not run on hardware
//
#include <hip/hip_runtime.h>
#include <hip/hip_bf16.h>

typedef __bf16 bf16;
typedef __attribute__((ext_vector_type(16))) __bf16 v16bf;
typedef __attribute__((ext_vector_type(8)))  float  v8f;

#define BATCH 16
#define SEQ 1024
#define DIMX 512
#define KEY_DIM 64
#define NUM_HEADS 8
#define VAL_DIM 128
#define QKV_OUT 2048
#define VAL_ATTN_DIM 1024
#define SCALE_F 0.125f

union CFrag { v8f v; float f[8]; };
union FragBits { v16bf v; uint4 q[2]; };

__device__ __forceinline__ v8f wmma_bf16(v16bf a, v16bf b, v8f c) {
  return __builtin_amdgcn_wmma_f32_16x16x32_bf16(false, a, false, b, (short)0, c,
                                                 false, false);
}

// One 16-byte async copy global -> LDS per lane (tracked on ASYNCcnt).
// Low 32 bits of a generic __shared__ pointer are the wave LDS byte offset.
__device__ __forceinline__ void async_copy_b128(const bf16* g, const bf16* l) {
  unsigned lds = (unsigned)(size_t)l;
  asm volatile("global_load_async_to_lds_b128 %0, %1, off"
               :: "v"(lds), "v"(g) : "memory");
}
__device__ __forceinline__ void wait_async0() {
  asm volatile("s_wait_asynccnt 0x0" ::: "memory");
}

// A-matrix 16x32 (MxK), row-major source, lda in elements.
// Lanes 0-15: row=lane, K={0..7,16..23}; lanes 16-31: row=lane-16, K={8..15,24..31}.
__device__ __forceinline__ v16bf load_frag_a(const bf16* __restrict__ p, int lda) {
  const int lane = threadIdx.x & 31;
  const int row  = lane & 15;
  const int koff = (lane >> 4) * 8;
  const bf16* r = p + row * lda + koff;
  FragBits f;
  f.q[0] = *(const uint4*)r;          // K koff..koff+7
  f.q[1] = *(const uint4*)(r + 16);   // K koff+16..koff+23
  return f.v;
}

// B-matrix 32x16 (KxN) supplied as B^T row-major (pt[n][k]), ldbt in elements.
// Lanes 0-15: col=lane, K={0..7,16..23}; lanes 16-31: col=lane-16, K={8..15,24..31}.
__device__ __forceinline__ v16bf load_frag_b(const bf16* __restrict__ pt, int ldbt) {
  const int lane = threadIdx.x & 31;
  const int col  = lane & 15;
  const int koff = (lane >> 4) * 8;
  const bf16* r = pt + col * ldbt + koff;
  FragBits f;
  f.q[0] = *(const uint4*)r;
  f.q[1] = *(const uint4*)(r + 16);
  return f.v;
}

// ---------------- conversion kernels ----------------
__global__ __launch_bounds__(256) void cvt_bf16(const float* __restrict__ s,
                                                bf16* __restrict__ d, int n) {
  int i = blockIdx.x * blockDim.x + threadIdx.x;
  if (i < n) d[i] = (bf16)s[i];
}

// d is [rows x cols] row-major, d[r][c] = s[c*rows + r]  (transpose+convert)
__global__ __launch_bounds__(256) void cvt_bf16_t(const float* __restrict__ s,
                                                  bf16* __restrict__ d,
                                                  int rows, int cols) {
  int i = blockIdx.x * blockDim.x + threadIdx.x;
  if (i < rows * cols) {
    int r = i / cols, c = i % cols;
    d[i] = (bf16)s[(size_t)c * rows + r];
  }
}

// ---------------- shared GEMM mainloop (block tile 64M x 128N, LDS double-buffer) ---
// sA: [2][64*32] bf16 (4KB/stage), sB: [2][128*32] bf16 (8KB/stage).
// c[2][2] accumulators for this wave's 32x32 tile.
template <int K>
__device__ __forceinline__ void gemm_mainloop(const bf16* __restrict__ A,
                                              const bf16* __restrict__ Bt,
                                              int m0blk, int n0blk,
                                              bf16* sA, bf16* sB, CFrag c[2][2]) {
  const int tid = threadIdx.x;
  const int w = tid >> 5;
  const int wm = w & 1, wn = w >> 1;
  const int crow = tid >> 2;            // 0..63   (A copy row; B rows crow, crow+64)
  const int cseg = (tid & 3) * 8;       // 16B segment within 32-element row

  const bf16* gA  = A  + (size_t)(m0blk + crow) * K + cseg;
  const bf16* gB0 = Bt + (size_t)(n0blk + crow) * K + cseg;
  const bf16* gB1 = Bt + (size_t)(n0blk + crow + 64) * K + cseg;
  bf16* lA  = sA + crow * 32 + cseg;
  bf16* lB0 = sB + crow * 32 + cseg;
  bf16* lB1 = sB + (crow + 64) * 32 + cseg;

  // prologue: stage 0
  async_copy_b128(gA, lA);
  async_copy_b128(gB0, lB0);
  async_copy_b128(gB1, lB1);

  constexpr int NS = K / 32;
#pragma unroll 1
  for (int s = 0; s < NS; ++s) {
    const int st = s & 1;
    wait_async0();          // own stage-s copies done (only stage in flight)
    __syncthreads();        // everyone's stage-s copies visible; stage s-1 reads done
    if (s + 1 < NS) {       // overlap stage s+1 copies with stage-s WMMAs
      const int nst = st ^ 1;
      const int kc = (s + 1) * 32;
      async_copy_b128(gA + kc, lA + nst * (64 * 32));
      async_copy_b128(gB0 + kc, lB0 + nst * (128 * 32));
      async_copy_b128(gB1 + kc, lB1 + nst * (128 * 32));
    }
    const bf16* pa = sA + st * (64 * 32) + (wm * 32) * 32;
    const bf16* pb = sB + st * (128 * 32) + (wn * 32) * 32;
    v16bf a0 = load_frag_a(pa, 32);
    v16bf a1 = load_frag_a(pa + 16 * 32, 32);
    v16bf b0 = load_frag_b(pb, 32);
    v16bf b1 = load_frag_b(pb + 16 * 32, 32);
    c[0][0].v = wmma_bf16(a0, b0, c[0][0].v);
    c[0][1].v = wmma_bf16(a0, b1, c[0][1].v);
    c[1][0].v = wmma_bf16(a1, b0, c[1][0].v);
    c[1][1].v = wmma_bf16(a1, b1, c[1][1].v);
  }
}

// ---------------- QKV GEMM: [16384,512]x[512,2048] -> scatter q/k/v^T ----------------
__global__ __launch_bounds__(256) void qkv_gemm(const bf16* __restrict__ A,
                                                const bf16* __restrict__ Bt,
                                                const float* __restrict__ bias,
                                                bf16* __restrict__ qd,
                                                bf16* __restrict__ kd,
                                                bf16* __restrict__ vtd) {
  __shared__ __attribute__((aligned(16))) bf16 sA[2][64 * 32];
  __shared__ __attribute__((aligned(16))) bf16 sB[2][128 * 32];
  const int w = threadIdx.x >> 5, lane = threadIdx.x & 31;
  const int wm = w & 1, wn = w >> 1;
  const int m0blk = blockIdx.y * 64, n0blk = blockIdx.x * 128;
  const int m0 = m0blk + wm * 32, n0 = n0blk + wn * 32;
  CFrag c[2][2] = {};
  gemm_mainloop<DIMX>(A, Bt, m0blk, n0blk, &sA[0][0], &sB[0][0], c);

  const int rowb = (lane >> 4) * 8, col = lane & 15;
#pragma unroll
  for (int j = 0; j < 2; ++j) {
    const int gn = n0 + j * 16 + col;
    const float bj = bias[gn];
    const int h = gn >> 8, cc = gn & 255;
#pragma unroll
    for (int i = 0; i < 2; ++i) {
#pragma unroll
      for (int r = 0; r < 8; ++r) {
        const int gm = m0 + i * 16 + rowb + r;
        const int bb = gm >> 10, tt = gm & 1023;
        const bf16 bv = (bf16)(c[i][j].f[r] + bj);
        const size_t base = (size_t)(bb * NUM_HEADS + h);
        if (cc < 64)
          qd[(base * SEQ + tt) * KEY_DIM + cc] = bv;
        else if (cc < 128)
          kd[(base * SEQ + tt) * KEY_DIM + (cc - 64)] = bv;
        else
          vtd[(base * VAL_DIM + (cc - 128)) * SEQ + tt] = bv;
      }
    }
  }
}

// ---------------- proj GEMM: [16384,1024]x[1024,512]+b -> fp32 out ----------------
__global__ __launch_bounds__(256) void proj_gemm(const bf16* __restrict__ A,
                                                 const bf16* __restrict__ Bt,
                                                 const float* __restrict__ bias,
                                                 float* __restrict__ out) {
  __shared__ __attribute__((aligned(16))) bf16 sA[2][64 * 32];
  __shared__ __attribute__((aligned(16))) bf16 sB[2][128 * 32];
  const int w = threadIdx.x >> 5, lane = threadIdx.x & 31;
  const int wm = w & 1, wn = w >> 1;
  const int m0blk = blockIdx.y * 64, n0blk = blockIdx.x * 128;
  const int m0 = m0blk + wm * 32, n0 = n0blk + wn * 32;
  CFrag c[2][2] = {};
  gemm_mainloop<VAL_ATTN_DIM>(A, Bt, m0blk, n0blk, &sA[0][0], &sB[0][0], c);

  const int rowb = (lane >> 4) * 8, col = lane & 15;
#pragma unroll
  for (int j = 0; j < 2; ++j) {
    const int gn = n0 + j * 16 + col;
    const float bj = bias[gn];
#pragma unroll
    for (int i = 0; i < 2; ++i) {
#pragma unroll
      for (int r = 0; r < 8; ++r) {
        const int gm = m0 + i * 16 + rowb + r;
        out[(size_t)gm * DIMX + gn] = c[i][j].f[r] + bj;
      }
    }
  }
}

// ---------------- fused attention: one block per (b, h, 16-query tile) ----------------
__global__ __launch_bounds__(256) void attn_kernel(const bf16* __restrict__ q,
                                                   const bf16* __restrict__ k,
                                                   const bf16* __restrict__ vt,
                                                   const float* __restrict__ attn_biases,
                                                   const int* __restrict__ bias_idxs,
                                                   bf16* __restrict__ aout) {
  __shared__ __attribute__((aligned(16))) float logits[16][SEQ];  // 64 KB
  __shared__ __attribute__((aligned(16))) bf16  probs[16][SEQ];   // 32 KB
  __shared__ float red[16][16];
  __shared__ float rowmax[16];
  __shared__ float rowinv[16];

  const int qb = blockIdx.x, h = blockIdx.y, b = blockIdx.z;
  const int q0 = qb * 16;
  const int w = threadIdx.x >> 5, lane = threadIdx.x & 31;
  const size_t bh = (size_t)(b * NUM_HEADS + h);
  const bf16* qbh = q + bh * SEQ * KEY_DIM;
  const bf16* kbh = k + bh * SEQ * KEY_DIM;
  const bf16* vbh = vt + bh * VAL_DIM * SEQ;
  const int rowb = (lane >> 4) * 8, colL = lane & 15;

  // ---- Phase A: logits = q k^T * scale + bias, 8 key tiles per wave ----
  const v16bf aq0 = load_frag_a(qbh + (size_t)q0 * KEY_DIM + 0, KEY_DIM);
  const v16bf aq1 = load_frag_a(qbh + (size_t)q0 * KEY_DIM + 32, KEY_DIM);
  for (int kt = w * 8; kt < w * 8 + 8; ++kt) {
    v16bf bk0 = load_frag_b(kbh + (size_t)kt * 16 * KEY_DIM + 0, KEY_DIM);
    v16bf bk1 = load_frag_b(kbh + (size_t)kt * 16 * KEY_DIM + 32, KEY_DIM);
    CFrag c = {};
    c.v = wmma_bf16(aq0, bk0, c.v);
    c.v = wmma_bf16(aq1, bk1, c.v);
    const int kj = kt * 16 + colL;
#pragma unroll
    for (int r = 0; r < 8; ++r) {
      const int qi = q0 + rowb + r;
      const int idx = bias_idxs[qi * SEQ + kj];
      logits[rowb + r][kj] = c.f[r] * SCALE_F + attn_biases[h * SEQ + idx];
    }
  }
  __syncthreads();

  // ---- Phase B: softmax (16 rows x 16 threads each) ----
  {
    const int t = threadIdx.x, row = t >> 4, lc = t & 15;
    float m = -3.0e38f;
    for (int c = lc; c < SEQ; c += 16) m = fmaxf(m, logits[row][c]);
    red[row][lc] = m;
    __syncthreads();
    if (lc == 0) {
      float mm = red[row][0];
      for (int j = 1; j < 16; ++j) mm = fmaxf(mm, red[row][j]);
      rowmax[row] = mm;
    }
    __syncthreads();
    const float mm = rowmax[row];
    float s = 0.f;
    for (int c = lc; c < SEQ; c += 16) {
      float p = __expf(logits[row][c] - mm);
      probs[row][c] = (bf16)p;   // unnormalized; 1/sum folded into epilogue
      s += p;
    }
    red[row][lc] = s;
    __syncthreads();
    if (lc == 0) {
      float ss = 0.f;
      for (int j = 0; j < 16; ++j) ss += red[row][j];
      rowinv[row] = 1.0f / ss;
    }
    __syncthreads();
  }

  // ---- Phase C: out = probs @ v, each wave owns 16 v-columns ----
  const int col0 = w * 16;
  CFrag o = {};
  for (int kc = 0; kc < SEQ; kc += 32) {
    v16bf ap = load_frag_a(&probs[0][0] + kc, SEQ);           // ds loads
    v16bf bv = load_frag_b(vbh + (size_t)col0 * SEQ + kc, SEQ);
    o.v = wmma_bf16(ap, bv, o.v);
  }
  const int tokBase = b * SEQ + q0;
#pragma unroll
  for (int r = 0; r < 8; ++r) {
    const int row = rowb + r;
    const float val = o.f[r] * rowinv[row];
    aout[(size_t)(tokBase + row) * VAL_ATTN_DIM + h * VAL_DIM + col0 + colL] = (bf16)val;
  }
}

extern "C" void kernel_launch(void* const* d_in, const int* in_sizes, int n_in,
                              void* d_out, int out_size, void* d_ws, size_t ws_size,
                              hipStream_t stream) {
  const float* x          = (const float*)d_in[0];
  const float* qkv_w      = (const float*)d_in[1];
  const float* qkv_b      = (const float*)d_in[2];
  const float* proj_w     = (const float*)d_in[3];
  const float* proj_b     = (const float*)d_in[4];
  const float* attn_bias  = (const float*)d_in[5];
  const int*   bias_idxs  = (const int*)d_in[6];
  float* out = (float*)d_out;

  const size_t TOK = (size_t)BATCH * SEQ;          // 16384
  bf16* xb     = (bf16*)d_ws;                      // [16384,512]
  bf16* wqkvt  = xb + TOK * DIMX;                  // [2048,512]   (qkv_w^T)
  bf16* wprojt = wqkvt + (size_t)QKV_OUT * DIMX;   // [512,1024]   (proj_w^T)
  bf16* qd     = wprojt + (size_t)DIMX * VAL_ATTN_DIM;               // [B,H,N,64]
  bf16* kd     = qd + (size_t)BATCH * NUM_HEADS * SEQ * KEY_DIM;     // [B,H,N,64]
  bf16* vtd    = kd + (size_t)BATCH * NUM_HEADS * SEQ * KEY_DIM;     // [B,H,128,N]
  bf16* aoutd  = vtd + (size_t)BATCH * NUM_HEADS * VAL_DIM * SEQ;    // [16384,1024]

  {
    int n = (int)(TOK * DIMX);
    cvt_bf16<<<(n + 255) / 256, 256, 0, stream>>>(x, xb, n);
  }
  cvt_bf16_t<<<(QKV_OUT * DIMX + 255) / 256, 256, 0, stream>>>(qkv_w, wqkvt,
                                                               QKV_OUT, DIMX);
  cvt_bf16_t<<<(DIMX * VAL_ATTN_DIM + 255) / 256, 256, 0, stream>>>(proj_w, wprojt,
                                                                    DIMX, VAL_ATTN_DIM);

  qkv_gemm<<<dim3(QKV_OUT / 128, TOK / 64), 256, 0, stream>>>(xb, wqkvt, qkv_b,
                                                              qd, kd, vtd);
  attn_kernel<<<dim3(SEQ / 16, NUM_HEADS, BATCH), 256, 0, stream>>>(
      qd, kd, vtd, attn_bias, bias_idxs, aoutd);
  proj_gemm<<<dim3(DIMX / 128, TOK / 64), 256, 0, stream>>>(aoutd, wprojt, proj_b,
                                                            out);
}
